// DualChannelGAT_29703993819788
// MI455X (gfx1250) — compile-verified, matrix operand order checked by reading, not run on hardware
//
#include <hip/hip_runtime.h>
#include <hip/hip_bf16.h>

// ---------------------------------------------------------------------------
// Types for CDNA5 WMMA
// ---------------------------------------------------------------------------
typedef __attribute__((ext_vector_type(16))) __bf16 v16bf;
typedef __attribute__((ext_vector_type(8)))  float  v8f;

union Frag16 { v16bf v; unsigned int u[8]; };

static __device__ __forceinline__ unsigned short f2bf(float f) {
    unsigned u = __builtin_bit_cast(unsigned, f);
    unsigned r = 0x7FFFu + ((u >> 16) & 1u);   // round to nearest even
    u += r;
    return (unsigned short)(u >> 16);
}
static __device__ __forceinline__ float leaky(float x) { return x > 0.0f ? x : 0.2f * x; }

// ---------------------------------------------------------------------------
// Generic tiled WMMA GEMM:  C[M,N] = epilogue( A[M,K] @ B[K,N] )
// Block tile 128x64, BK=32, 8 waves (wave32) in a 4x2 grid; each wave owns a
// 32x32 register tile = 2x2 WMMA accumulators -> 4 v_wmma per K-step between
// barriers. A/B staged in LDS as bf16; fragments per the CDNA5 16-bit layouts.
// M must be a multiple of 128; columns are guarded by Ncols.
// ---------------------------------------------------------------------------
enum { A_F32 = 0, A_BF16, A_ADJ, A_ADJ2, A_CONCAT };
enum { B_F32 = 0, B_BF16, B_WGATHER };
enum { E_BNLEAKY = 0, E_AGG, E_ADDRELU, E_SIGMOID };

struct GemmParams {
    const void* A;      // A matrix (f32 / bf16 / int32 adj / concat part 1)
    const void* A2;     // second adj / concat part 2
    const void* Bsrc;   // B matrix (f32 / bf16 / W[h,f,d] gather)
    float*      C;
    int  K;
    int  Ncols;                           // valid output columns (guard)
    int  lda, ldb, ldc;
    long strideA, strideB, strideC;       // per-batch element strides
    int  kSplit;                          // A_CONCAT split point
    int  wgF, wgD;                        // B_WGATHER dims (F, D)
    const float* gamma;                   // E_BNLEAKY scale source
    const float* beta;                    // BN shift / bias
    const float* base;                    // E_ADDRELU additive base (same layout as C)
    const float* ew;                      // E_AGG exp(e - m) factors
    long ewBatchStride;
    int  ewRowStride;
    int  nDivD;                           // col -> head divisor (D)
};

template <int AM, int BM, int EM>
__global__ __launch_bounds__(256) void wmma_gemm(GemmParams p) {
    __shared__ unsigned short As[128][36];  // [row][k], pitch 36 (even, padded)
    __shared__ unsigned short Bs[64][36];   // transposed: [n][k]

    const int tid  = threadIdx.x;
    const int lane = tid & 31;
    const int wave = tid >> 5;
    const int waveRow = wave >> 1;          // 0..3  (32 rows each)
    const int waveCol = wave & 1;           // 0..1  (32 cols each)
    const int bz = blockIdx.z;

    const long aB = (long)bz * p.strideA;
    const long bB = (long)bz * p.strideB;
    const long cB = (long)bz * p.strideC;
    const int rowBlock = blockIdx.y * 128;
    const int colBlock = blockIdx.x * 64;

    v8f acc[2][2];
    #pragma unroll
    for (int i = 0; i < 2; ++i)
        #pragma unroll
        for (int j = 0; j < 2; ++j)
            acc[i][j] = (v8f){0.f, 0.f, 0.f, 0.f, 0.f, 0.f, 0.f, 0.f};

    for (int k0 = 0; k0 < p.K; k0 += 32) {
        // ---- stage A tile (128 x 32) as bf16 ----
        #pragma unroll
        for (int it = 0; it < 16; ++it) {
            int idx = tid + it * 256;
            int r = idx >> 5, kk = idx & 31;
            int gr = rowBlock + r, gk = k0 + kk;
            unsigned short v;
            if (AM == A_F32) {
                const float* ap = (const float*)p.A + aB + (long)gr * p.lda + gk;
                v = f2bf(*ap);
                if (it == 0 && k0 + 32 < p.K) __builtin_prefetch(ap + 32, 0, 3);
            } else if (AM == A_BF16) {
                const unsigned short* ap =
                    (const unsigned short*)p.A + aB + (long)gr * p.lda + gk;
                v = *ap;
                if (it == 0 && k0 + 32 < p.K) __builtin_prefetch(ap + 32, 0, 3);
            } else if (AM == A_ADJ) {
                const int* ap = (const int*)p.A + aB + (long)gr * p.lda + gk;
                int a = *ap;
                v = (a > 0) ? (unsigned short)0x3F80 : (unsigned short)0;
                if (it == 0 && k0 + 32 < p.K) __builtin_prefetch(ap + 32, 0, 3);
            } else if (AM == A_ADJ2) {
                long o = aB + (long)gr * p.lda + gk;
                int a = ((const int*)p.A)[o] + ((const int*)p.A2)[o];
                v = (a > 0) ? (unsigned short)0x3F80 : (unsigned short)0;
            } else { // A_CONCAT: [f1 | f2] along K
                float f = (gk < p.kSplit)
                              ? ((const float*)p.A )[(long)gr * p.kSplit + gk]
                              : ((const float*)p.A2)[(long)gr * p.kSplit + (gk - p.kSplit)];
                v = f2bf(f);
            }
            As[r][kk] = v;
        }
        // ---- stage B tile (32 x 64) transposed, as bf16, column-guarded ----
        #pragma unroll
        for (int it = 0; it < 8; ++it) {
            int idx = tid + it * 256;
            int kk = idx >> 6, n = idx & 63;
            int gk = k0 + kk, gn = colBlock + n;
            unsigned short v = 0;
            if (gn < p.Ncols) {
                if (BM == B_F32) {
                    const float* bp = (const float*)p.Bsrc + bB + (long)gk * p.ldb + gn;
                    v = f2bf(*bp);
                    if (it == 0 && k0 + 32 < p.K)
                        __builtin_prefetch(bp + (long)32 * p.ldb, 0, 3);
                } else if (BM == B_BF16) {
                    const unsigned short* bp =
                        (const unsigned short*)p.Bsrc + bB + (long)gk * p.ldb + gn;
                    v = *bp;
                    if (it == 0 && k0 + 32 < p.K)
                        __builtin_prefetch(bp + (long)32 * p.ldb, 0, 3);
                } else { // B_WGATHER: W[h, f, d] -> column (h*D + d)
                    int hh = gn / p.wgD, dd = gn - hh * p.wgD;
                    v = f2bf(((const float*)p.Bsrc)[((long)hh * p.wgF + gk) * p.wgD + dd]);
                }
            }
            Bs[n][kk] = v;
        }
        __syncthreads();

        // ---- build 2 A-frags + 2 B-frags, issue 4 WMMAs ----
        Frag16 fa[2], fb[2];
        const int m0 = waveRow * 32 + (lane & 15);
        const int n0 = waveCol * 32 + (lane & 15);
        const int kh = (lane >> 4) << 3;   // 0 or 8
        #pragma unroll
        for (int t = 0; t < 2; ++t) {
            #pragma unroll
            for (int v2 = 0; v2 < 4; ++v2) {
                fa[t].u[v2]     = *(const unsigned int*)&As[m0 + 16 * t][kh + 2 * v2];
                fa[t].u[v2 + 4] = *(const unsigned int*)&As[m0 + 16 * t][16 + kh + 2 * v2];
                fb[t].u[v2]     = *(const unsigned int*)&Bs[n0 + 16 * t][kh + 2 * v2];
                fb[t].u[v2 + 4] = *(const unsigned int*)&Bs[n0 + 16 * t][16 + kh + 2 * v2];
            }
        }
        #pragma unroll
        for (int tr = 0; tr < 2; ++tr)
            #pragma unroll
            for (int tc = 0; tc < 2; ++tc)
                acc[tr][tc] = __builtin_amdgcn_wmma_f32_16x16x32_bf16(
                    false, fa[tr].v, false, fb[tc].v, (short)0, acc[tr][tc], false, false);
        __syncthreads();
    }

    // ---- epilogue: lane element v2 -> (m = v2 + 8*(lane>=16), n = lane&15) ----
    #pragma unroll
    for (int tc = 0; tc < 2; ++tc) {
        const int gcol = colBlock + waveCol * 32 + tc * 16 + (lane & 15);
        const bool colOK = gcol < p.Ncols;
        const int gc = colOK ? gcol : 0;
        float s0 = 0.f, s1 = 0.f;
        if (EM == E_BNLEAKY) { s0 = p.gamma[gc] * rsqrtf(1.0f + 1e-5f); s1 = p.beta[gc]; }
        if (EM == E_ADDRELU || EM == E_SIGMOID) { s1 = p.beta[gc]; }
        const float* ewp = nullptr;
        if (EM == E_AGG)
            ewp = p.ew + (long)bz * p.ewBatchStride + (long)(gc / p.nDivD) * p.ewRowStride;

        #pragma unroll
        for (int tr = 0; tr < 2; ++tr) {
            #pragma unroll
            for (int v2 = 0; v2 < 8; ++v2) {
                int mm   = v2 + ((lane >> 4) << 3);
                int grow = rowBlock + waveRow * 32 + tr * 16 + mm;
                long ci  = cB + (long)grow * p.ldc + gcol;
                float x = acc[tr][tc][v2], o;
                if (EM == E_BNLEAKY) {
                    o = leaky(x * s0 + s1);
                } else if (EM == E_AGG) {
                    o = leaky(x * ewp[grow]);
                } else if (EM == E_ADDRELU) {
                    float t = x + s1; t = t > 0.f ? t : 0.f;
                    o = p.base[(long)grow * p.ldc + gc] + t;
                } else { // E_SIGMOID
                    o = 1.0f / (1.0f + __expf(-(x + s1)));
                }
                if (colOK) p.C[ci] = o;
            }
        }
    }
}

// ---------------------------------------------------------------------------
// e[b,h,n] = leaky( h[b,n, h*D : h*D+D] . a[h,:] )
// ---------------------------------------------------------------------------
__global__ __launch_bounds__(256) void attn_e_kernel(
    const float* __restrict__ h, const float* __restrict__ a, float* __restrict__ e,
    int H, int N, int Dd, int ldh) {
    long idx = (long)blockIdx.x * 256 + threadIdx.x;       // over B*H*N
    int n = (int)(idx % N);
    long t = idx / N;
    int hh = (int)(t % H);
    long b = t / H;
    const float* row = h + (b * N + n) * (long)ldh + hh * Dd;
    const float* av  = a + hh * Dd;
    float s = 0.f;
    for (int d = 0; d < Dd; ++d) s += row[d] * av[d];
    e[idx] = leaky(s);
}

// m = max_n e(row); ew[n] = exp(e[n] - m).  One block per (b*H + h) row.
__global__ __launch_bounds__(256) void max_exp_kernel(
    const float* __restrict__ e, float* __restrict__ ew, int N) {
    __shared__ float red[256];
    const float* er = e + (long)blockIdx.x * N;
    float m = -1e30f;
    for (int i = threadIdx.x; i < N; i += 256) m = fmaxf(m, er[i]);
    red[threadIdx.x] = m;
    __syncthreads();
    for (int s = 128; s > 0; s >>= 1) {
        if (threadIdx.x < s) red[threadIdx.x] = fmaxf(red[threadIdx.x], red[threadIdx.x + s]);
        __syncthreads();
    }
    m = red[0];
    float* wr = ew + (long)blockIdx.x * N;
    for (int i = threadIdx.x; i < N; i += 256) wr[i] = __expf(er[i] - m);
}

// invden[b,h,j] = 1 / sum_i adj[b,i,j] * ew[b,h,i]   (column reduction, coalesced in j)
template <int H, bool TWO>
__global__ __launch_bounds__(256) void denom_kernel(
    const int* __restrict__ adj, const int* __restrict__ adj2,
    const float* __restrict__ ew, float* __restrict__ invden, int N) {
    __shared__ float ews[H * 512];
    const int b = blockIdx.y;
    for (int i = threadIdx.x; i < H * N; i += 256) ews[i] = ew[(long)b * H * N + i];
    __syncthreads();
    const int j = blockIdx.x * 256 + threadIdx.x;
    float acc[H];
    #pragma unroll
    for (int hh = 0; hh < H; ++hh) acc[hh] = 0.f;
    const long base = (long)b * N * N + j;
    for (int i = 0; i < N; ++i) {
        int a = adj[base + (long)i * N];
        if (TWO) a += adj2[base + (long)i * N];
        if (a > 0) {
            #pragma unroll
            for (int hh = 0; hh < H; ++hh) acc[hh] += ews[hh * N + i];
        }
    }
    #pragma unroll
    for (int hh = 0; hh < H; ++hh) {
        float d = acc[hh];
        invden[(long)b * H * N + hh * N + j] = d > 0.f ? 1.0f / d : 0.0f;
    }
}

// G[b,j,hd] = bf16( h[b,j,hd] * invden[b, hd/D, j] )
__global__ __launch_bounds__(256) void scale_bf16_kernel(
    const float* __restrict__ h, const float* __restrict__ invden,
    unsigned short* __restrict__ G, int N, int H, int Dd, long total) {
    long idx = (long)blockIdx.x * 256 + threadIdx.x;
    if (idx >= total) return;
    int HD = H * Dd;
    int c = (int)(idx % HD);
    long bn = idx / HD;
    int j = (int)(bn % N);
    long b = bn / N;
    float g = h[idx] * invden[(b * H + c / Dd) * (long)N + j];
    G[idx] = f2bf(g);
}

// u1 = gs*f1+(1-gs)*f2 ; u2 = go*f2+(1-go)*f1 ; xcat = bf16([u1 | u2])
__global__ __launch_bounds__(256) void gate_combine_kernel(
    const float* __restrict__ f1, const float* __restrict__ f2,
    const float* __restrict__ gs, const float* __restrict__ go_,
    unsigned short* __restrict__ xcat, long total /* rows*768 */) {
    long idx = (long)blockIdx.x * 256 + threadIdx.x;
    if (idx >= total) return;
    int c = (int)(idx % 768);
    long r = idx / 768;
    float a1 = f1[idx], a2 = f2[idx], s = gs[idx], o = go_[idx];
    float u1 = s * a1 + (1.0f - s) * a2;
    float u2 = o * a2 + (1.0f - o) * a1;
    xcat[r * 1536 + c]       = f2bf(u1);
    xcat[r * 1536 + 768 + c] = f2bf(u2);
}

// pooled[b,d] = mean_n xo[b,n,d]; logits = pooled @ wc + bc
__global__ __launch_bounds__(128) void pool_cls_kernel(
    const float* __restrict__ xo, const float* __restrict__ wc,
    const float* __restrict__ bc, float* __restrict__ out, int N, int Dd, int C) {
    __shared__ float pooled[128];
    const int b = blockIdx.x;
    const int d = threadIdx.x;
    float s = 0.f;
    if (d < Dd)
        for (int n = 0; n < N; ++n) s += xo[((long)b * N + n) * Dd + d];
    pooled[d] = s / (float)N;
    __syncthreads();
    if (d < C) {
        float l = bc[d];
        for (int k = 0; k < Dd; ++k) l += pooled[k] * wc[k * C + d];
        out[b * C + d] = l;
    }
}

// ---------------------------------------------------------------------------
// Host-side orchestration
// ---------------------------------------------------------------------------
extern "C" void kernel_launch(void* const* d_in, const int* in_sizes, int n_in,
                              void* d_out, int out_size, void* d_ws, size_t ws_size,
                              hipStream_t stream) {
    (void)in_sizes; (void)n_in; (void)out_size; (void)ws_size;
    constexpr int B = 16, N = 512, F = 768, H = 8, D = 96;
    constexpr int HD = H * D;            // 768
    constexpr int HD2 = 2 * HD;          // 1536
    constexpr int M = B * N;             // 8192

    const float* fea  = (const float*)d_in[0];
    const int*   adj1 = (const int*)  d_in[1];
    const int*   adj2 = (const int*)  d_in[2];
    const float* w1   = (const float*)d_in[3];
    const float* a1   = (const float*)d_in[4];
    const float* g1   = (const float*)d_in[5];
    const float* b1   = (const float*)d_in[6];
    const float* w2   = (const float*)d_in[7];
    const float* a2   = (const float*)d_in[8];
    const float* g2   = (const float*)d_in[9];
    const float* b2   = (const float*)d_in[10];
    const float* wt   = (const float*)d_in[11];
    const float* bt   = (const float*)d_in[12];
    const float* wgs  = (const float*)d_in[13];
    const float* bgs  = (const float*)d_in[14];
    const float* wgo  = (const float*)d_in[15];
    const float* bgo  = (const float*)d_in[16];
    const float* wo   = (const float*)d_in[17];
    const float* ao   = (const float*)d_in[18];
    const float* go_g = (const float*)d_in[19];
    const float* bo   = (const float*)d_in[20];
    const float* wc   = (const float*)d_in[21];
    const float* bc   = (const float*)d_in[22];

    // workspace carve-up
    char* ws = (char*)d_ws;
    size_t off = 0;
    auto alloc = [&](size_t bytes) -> void* {
        void* p = ws + off;
        off = (off + bytes + 255) & ~(size_t)255;
        return p;
    };
    float*          hbuf    = (float*)         alloc((size_t)M * HD * 4);   // h, later f1
    unsigned short* Gbf     = (unsigned short*)alloc((size_t)M * HD * 2);
    float*          x1      = (float*)         alloc((size_t)M * HD * 4);
    float*          x2      = (float*)         alloc((size_t)M * HD * 4);
    float*          f2buf   = (float*)         alloc((size_t)M * HD * 4);
    float*          gsBuf   = (float*)         alloc((size_t)M * HD * 4);
    float*          goBuf   = (float*)         alloc((size_t)M * HD * 4);
    unsigned short* xcat    = (unsigned short*)alloc((size_t)M * HD2 * 2);
    float*          ho      = (float*)         alloc((size_t)M * D * 4);
    float*          xo      = (float*)         alloc((size_t)M * D * 4);
    float*          ebuf    = (float*)         alloc((size_t)B * H * N * 4);
    float*          ewbuf   = (float*)         alloc((size_t)B * H * N * 4);
    float*          invden  = (float*)         alloc((size_t)B * H * N * 4);
    float*          eobuf   = (float*)         alloc((size_t)B * N * 4);
    float*          ewobuf  = (float*)         alloc((size_t)B * N * 4);
    float*          invdeno = (float*)         alloc((size_t)B * N * 4);
    unsigned short* Gobf    = (unsigned short*)alloc((size_t)M * D * 2);

    // ---- one GAT channel: proj GEMM -> e -> softmax pieces -> masked agg GEMM
    auto run_channel = [&](const float* W, const float* av, const float* gamma,
                           const float* beta, const int* adj, float* xout) {
        GemmParams pp{};
        pp.A = fea; pp.Bsrc = W; pp.C = hbuf;
        pp.K = F; pp.Ncols = HD; pp.lda = F; pp.ldb = 0; pp.ldc = HD;
        pp.wgF = F; pp.wgD = D;
        pp.gamma = gamma; pp.beta = beta;
        wmma_gemm<A_F32, B_WGATHER, E_BNLEAKY>
            <<<dim3(HD / 64, M / 128, 1), 256, 0, stream>>>(pp);

        attn_e_kernel<<<(B * H * N) / 256, 256, 0, stream>>>(hbuf, av, ebuf, H, N, D, HD);
        max_exp_kernel<<<B * H, 256, 0, stream>>>(ebuf, ewbuf, N);
        denom_kernel<H, false><<<dim3(N / 256, B), 256, 0, stream>>>(adj, nullptr, ewbuf, invden, N);
        scale_bf16_kernel<<<(unsigned)(((size_t)M * HD + 255) / 256), 256, 0, stream>>>(
            hbuf, invden, Gbf, N, H, D, (long)M * HD);

        GemmParams pa{};
        pa.A = adj; pa.Bsrc = Gbf; pa.C = xout;
        pa.K = N; pa.Ncols = HD; pa.lda = N; pa.ldb = HD; pa.ldc = HD;
        pa.strideA = (long)N * N; pa.strideB = (long)N * HD; pa.strideC = (long)N * HD;
        pa.ew = ewbuf; pa.ewBatchStride = (long)H * N; pa.ewRowStride = N; pa.nDivD = D;
        wmma_gemm<A_ADJ, B_BF16, E_AGG>
            <<<dim3(HD / 64, N / 128, B), 256, 0, stream>>>(pa);
    };

    run_channel(w1, a1, g1, b1, adj1, x1);
    run_channel(w2, a2, g2, b2, adj2, x2);

    // ---- gated interaction: f1 = x1 + relu(x2@wt+bt); f2 = x2 + relu(x1@wt+bt)
    {
        GemmParams pf{};
        pf.A = x2; pf.Bsrc = wt; pf.C = hbuf;        // reuse hbuf as f1
        pf.K = HD; pf.Ncols = HD; pf.lda = HD; pf.ldb = HD; pf.ldc = HD;
        pf.beta = bt; pf.base = x1;
        wmma_gemm<A_F32, B_F32, E_ADDRELU>
            <<<dim3(HD / 64, M / 128, 1), 256, 0, stream>>>(pf);
        pf.A = x1; pf.C = f2buf; pf.base = x2;
        wmma_gemm<A_F32, B_F32, E_ADDRELU>
            <<<dim3(HD / 64, M / 128, 1), 256, 0, stream>>>(pf);
    }
    // gs = sigmoid([f1|f2]@wgs+bgs); go = sigmoid([f1|f2]@wgo+bgo)
    {
        GemmParams pg{};
        pg.A = hbuf; pg.A2 = f2buf; pg.kSplit = HD;
        pg.Bsrc = wgs; pg.C = gsBuf;
        pg.K = HD2; pg.Ncols = HD; pg.ldb = HD; pg.ldc = HD; pg.beta = bgs;
        wmma_gemm<A_CONCAT, B_F32, E_SIGMOID>
            <<<dim3(HD / 64, M / 128, 1), 256, 0, stream>>>(pg);
        pg.Bsrc = wgo; pg.C = goBuf; pg.beta = bgo;
        wmma_gemm<A_CONCAT, B_F32, E_SIGMOID>
            <<<dim3(HD / 64, M / 128, 1), 256, 0, stream>>>(pg);
    }
    gate_combine_kernel<<<(unsigned)(((size_t)M * HD + 255) / 256), 256, 0, stream>>>(
        hbuf, f2buf, gsBuf, goBuf, xcat, (long)M * HD);

    // ---- out-attention head: ho = leaky(BN(xcat @ wo))  (N=96, guarded)
    {
        GemmParams po{};
        po.A = xcat; po.Bsrc = wo; po.C = ho;
        po.K = HD2; po.Ncols = D; po.lda = HD2; po.ldb = D; po.ldc = D;
        po.gamma = go_g; po.beta = bo;
        wmma_gemm<A_BF16, B_F32, E_BNLEAKY>
            <<<dim3(2, M / 128, 1), 256, 0, stream>>>(po);
    }
    attn_e_kernel<<<(B * N) / 256, 256, 0, stream>>>(ho, ao, eobuf, 1, N, D, D);
    max_exp_kernel<<<B, 256, 0, stream>>>(eobuf, ewobuf, N);
    denom_kernel<1, true><<<dim3(N / 256, B), 256, 0, stream>>>(adj1, adj2, ewobuf, invdeno, N);
    scale_bf16_kernel<<<(unsigned)(((size_t)M * D + 255) / 256), 256, 0, stream>>>(
        ho, invdeno, Gobf, N, 1, D, (long)M * D);
    {
        GemmParams pq{};
        pq.A = adj1; pq.A2 = adj2; pq.Bsrc = Gobf; pq.C = xo;
        pq.K = N; pq.Ncols = D; pq.lda = N; pq.ldb = D; pq.ldc = D;
        pq.strideA = (long)N * N; pq.strideB = (long)N * D; pq.strideC = (long)N * D;
        pq.ew = ewobuf; pq.ewBatchStride = N; pq.ewRowStride = N; pq.nDivD = D;
        wmma_gemm<A_ADJ2, B_BF16, E_AGG>
            <<<dim3(2, N / 128, B), 256, 0, stream>>>(pq);
    }

    // ---- pool over nodes + classifier -> [B, 2]
    pool_cls_kernel<<<B, 128, 0, stream>>>(xo, wc, bc, (float*)d_out, N, D, 2);
}
